// Prior_Sp_7662221656260
// MI455X (gfx1250) — matrix operand order repaired and checked by reading, not compiled, not run on hardware
//
#include <hip/hip_runtime.h>

typedef unsigned short u16;
typedef unsigned int   u32;
typedef __attribute__((ext_vector_type(16))) __bf16 bf16x16;
typedef __attribute__((ext_vector_type(8)))  float  floatx8;

#define CDIM   192
#define C3     48
#define HEADS  2
#define CHH    96
#define HW     16384
#define BATCH  8
#define NTOT   (BATCH * HW)   // 131072

__device__ __forceinline__ u16 f2bf(float f) {
    union { float f; u32 u; } c; c.f = f;
    u32 r = c.u + 0x7FFFu + ((c.u >> 16) & 1u);
    return (u16)(r >> 16);
}
__device__ __forceinline__ float bf2f(u16 h) {
    union { u32 u; float f; } c; c.u = ((u32)h) << 16;
    return c.f;
}

union FragU { bf16x16 v; uint4 q[2]; };

// ---------------------------------------------------------------------------
// bf16 WMMA GEMM, one wave computes MT vertically-stacked 16x16 tiles that
// share a single B fragment (register-level operand reuse).
//   C[m][n] = sum_k A[m][k] * B_act[n][k]        (both operands K-contiguous)
// A element (m,k) at A[m*lda + k] (bf16), B element (k,n) at B[n*ldb + k].
// Batched over blockIdx.z = b*2 + h with independent (b,h) strides.
// EPI: 0 = transposed bf16 store   O[(ob*M + m)*ldo + n%nPerB]   (Q, K)
//      1 = channel-last bf16 store O[n*ldo + m]                  (V, attn@v)
//      2 = row-major f32 store     O[m*ldo + n]                  (logits)
//      3 = transposed f32 store    (final NCHW output)
// ---------------------------------------------------------------------------
template <int EPI, int MT>
__global__ __launch_bounds__(32) void gemm_wmma(
    const u16* __restrict__ A, const u16* __restrict__ B, void* __restrict__ O,
    int M, int K, int lda, int ldb, int ldo, int nPerB,
    size_t saB, size_t saH, size_t sbB, size_t sbH, size_t soB, size_t soH)
{
    const int z  = blockIdx.z;
    const int hb = z & 1;
    const int bb = z >> 1;
    A += (size_t)bb * saB + (size_t)hb * saH;
    B += (size_t)bb * sbB + (size_t)hb * sbH;
    const size_t oOff = (size_t)bb * soB + (size_t)hb * soH;

    const int lane = threadIdx.x & 31;
    const int hl   = lane >> 4;      // wave half
    const int lm   = lane & 15;
    const int mBase = blockIdx.y * (16 * MT);
    const int nBase = blockIdx.x * 16;

    // A: lanes 0-15 M=0..15, VGPR0-3 K=0..7, VGPR4-7 K=16..23 (+8 for upper half)
    const u16* aRow = A + (size_t)(mBase + lm) * lda + hl * 8;
    // B: lanes 0-15 N=0..15, VGPR0-3 K=0..7, VGPR4-7 K=16..23 (+8 for upper half)
    const u16* bRow = B + (size_t)(nBase + lm) * ldb + hl * 8;

    floatx8 acc[MT];
    #pragma unroll
    for (int t = 0; t < MT; ++t)
        acc[t] = (floatx8){0.f, 0.f, 0.f, 0.f, 0.f, 0.f, 0.f, 0.f};

    for (int kb = 0; kb < K; kb += 32) {
        FragU fb;
        fb.q[0] = *(const uint4*)(bRow);
        fb.q[1] = *(const uint4*)(bRow + 16);
        __builtin_prefetch(bRow + 8 * 32, 0, 3);   // WGP-scope global_prefetch_b8
        #pragma unroll
        for (int t = 0; t < MT; ++t) {
            FragU fa;
            const u16* ap = aRow + (size_t)t * 16 * lda;
            fa.q[0] = *(const uint4*)(ap);
            fa.q[1] = *(const uint4*)(ap + 16);
            acc[t] = __builtin_amdgcn_wmma_f32_16x16x32_bf16(
                false, fa.v, false, fb.v, (short)0, acc[t], false, false);
        }
        aRow += 32;
        bRow += 32;
    }

    // D layout: lane = N (lm), VGPR r -> M = tileBase + hl*8 + r
    const int n = nBase + lm;
    int ob = 0, nn = n;
    if ((EPI == 0 || EPI == 3) && nPerB) { ob = n / nPerB; nn = n % nPerB; }

    #pragma unroll
    for (int t = 0; t < MT; ++t) {
        const int m0 = mBase + t * 16 + hl * 8;
        if (EPI == 0) {                       // transposed bf16
            u16* o = (u16*)O + oOff;
            #pragma unroll
            for (int r = 0; r < 8; ++r)
                o[((size_t)ob * M + m0 + r) * (size_t)ldo + nn] = f2bf(acc[t][r]);
        } else if (EPI == 3) {                // transposed f32
            float* o = (float*)O + oOff;
            #pragma unroll
            for (int r = 0; r < 8; ++r)
                o[((size_t)ob * M + m0 + r) * (size_t)ldo + nn] = acc[t][r];
        } else if (EPI == 1) {                // channel-last bf16, packed 16B store
            u16* o = (u16*)O + oOff;
            uint4 pk;
            pk.x = (u32)f2bf(acc[t][0]) | ((u32)f2bf(acc[t][1]) << 16);
            pk.y = (u32)f2bf(acc[t][2]) | ((u32)f2bf(acc[t][3]) << 16);
            pk.z = (u32)f2bf(acc[t][4]) | ((u32)f2bf(acc[t][5]) << 16);
            pk.w = (u32)f2bf(acc[t][6]) | ((u32)f2bf(acc[t][7]) << 16);
            *(uint4*)(o + (size_t)n * ldo + m0) = pk;
        } else {                              // EPI == 2: row-major f32 (logits)
            float* o = (float*)O + oOff;
            #pragma unroll
            for (int r = 0; r < 8; ++r)
                o[(size_t)(m0 + r) * ldo + n] = acc[t][r];
        }
    }
}

// ---------------------------------------------------------------------------
// Partial 3x3 conv on first 48 channels + passthrough; NCHW fp32 in,
// channel-last bf16 out: T[(b*HW + n)*192 + oc]
// ---------------------------------------------------------------------------
__global__ __launch_bounds__(256) void pconv_pack(
    const float* __restrict__ X, const float* __restrict__ Wc, u16* __restrict__ T)
{
    size_t idx = (size_t)blockIdx.x * 256 + threadIdx.x;   // B*HW*192 threads
    int oc = (int)(idx % CDIM);
    size_t pix = idx / CDIM;          // b*HW + n
    int n = (int)(pix & (HW - 1));
    int b = (int)(pix >> 14);
    float r;
    if (oc < C3) {
        int y = n >> 7, x = n & 127;
        float acc = 0.f;
        for (int ic = 0; ic < C3; ++ic) {
            const float* xp = X + (((size_t)b * CDIM + ic) << 14);
            const float* wp = Wc + (size_t)(oc * C3 + ic) * 9;
            #pragma unroll
            for (int dy = 0; dy < 3; ++dy) {
                int yy = y + dy - 1;
                if ((unsigned)yy < 128u) {
                    #pragma unroll
                    for (int dx = 0; dx < 3; ++dx) {
                        int xx = x + dx - 1;
                        if ((unsigned)xx < 128u)
                            acc += wp[dy * 3 + dx] * xp[(yy << 7) + xx];
                    }
                }
            }
        }
        r = acc;
    } else {
        r = X[(((size_t)b * CDIM + oc) << 14) + n];
    }
    T[idx] = f2bf(r);
}

// 1/||row||_2 over 16384 contiguous bf16 per row
__global__ __launch_bounds__(256) void rownorm(const u16* __restrict__ Q,
                                               float* __restrict__ inv)
{
    __shared__ float red[256];
    const u16* p = Q + (size_t)blockIdx.x * HW;
    float s = 0.f;
    for (int i = threadIdx.x; i < HW; i += 256) {
        float f = bf2f(p[i]);
        s += f * f;
    }
    red[threadIdx.x] = s; __syncthreads();
    for (int st = 128; st > 0; st >>= 1) {
        if ((int)threadIdx.x < st) red[threadIdx.x] += red[threadIdx.x + st];
        __syncthreads();
    }
    if (threadIdx.x == 0) inv[blockIdx.x] = rsqrtf(red[0]);
}

// softmax over d of logits * qinv[c] * kinv[d] * temperature[h], bf16 out
__global__ __launch_bounds__(128) void softmax_rows(
    const float* __restrict__ L, const float* __restrict__ qinv,
    const float* __restrict__ kinv, const float* __restrict__ temp,
    u16* __restrict__ Aout)
{
    __shared__ float red[128];
    int row = blockIdx.x;            // (b*2+h)*96 + c
    int c  = row % CHH;
    int bh = row / CHH;
    int h  = bh & 1;
    int b  = bh >> 1;
    const float* lr = L + (size_t)bh * (CHH * CHH) + (size_t)c * CHH;
    float qi = qinv[b * CDIM + h * CHH + c];
    float tp = temp[h];
    int d = threadIdx.x;
    float val = 0.f, mval = -3.4e38f;
    if (d < CHH) {
        val = lr[d] * qi * kinv[b * CDIM + h * CHH + d] * tp;
        mval = val;
    }
    red[threadIdx.x] = mval; __syncthreads();
    for (int st = 64; st > 0; st >>= 1) {
        if ((int)threadIdx.x < st)
            red[threadIdx.x] = fmaxf(red[threadIdx.x], red[threadIdx.x + st]);
        __syncthreads();
    }
    float mx = red[0]; __syncthreads();
    float e = (d < CHH) ? __expf(val - mx) : 0.f;
    red[threadIdx.x] = e; __syncthreads();
    for (int st = 64; st > 0; st >>= 1) {
        if ((int)threadIdx.x < st) red[threadIdx.x] += red[threadIdx.x + st];
        __syncthreads();
    }
    float s = red[0];
    if (d < CHH)
        Aout[(size_t)bh * (CHH * CHH) + (size_t)c * CHH + d] = f2bf(e / s);
}

__global__ __launch_bounds__(256) void cvt_bf16(const float* __restrict__ in,
                                                u16* __restrict__ out, int n)
{
    int i = blockIdx.x * 256 + threadIdx.x;
    if (i < n) out[i] = f2bf(in[i]);
}

extern "C" void kernel_launch(void* const* d_in, const int* in_sizes, int n_in,
                              void* d_out, int out_size, void* d_ws, size_t ws_size,
                              hipStream_t stream)
{
    const float* x       = (const float*)d_in[0];
    const float* res     = (const float*)d_in[1];
    const float* w_qv_pc = (const float*)d_in[2];
    const float* w_qv    = (const float*)d_in[3];
    const float* w_k_pc  = (const float*)d_in[4];
    const float* w_k     = (const float*)d_in[5];
    const float* w_proj  = (const float*)d_in[6];
    const float* temp    = (const float*)d_in[7];
    float* out = (float*)d_out;

    const size_t ACT = (size_t)NTOT * CDIM;          // elements per activation tensor
    char* w = (char*)d_ws;
    u16*   t_x      = (u16*)(w + 0 * ACT);           // also reused as attn@v output
    u16*   t_res    = (u16*)(w + 2 * ACT);
    u16*   Qb       = (u16*)(w + 4 * ACT);           // [b*192][16384] bf16
    u16*   Kb       = (u16*)(w + 6 * ACT);
    u16*   Vb       = (u16*)(w + 8 * ACT);           // [b*n][192] bf16
    float* logits   = (float*)(w + 10 * ACT);                 // 16*96*96 f32
    u16*   attnb    = (u16*)(w + 10 * ACT + 589824);          // 16*96*96 bf16
    float* qinv     = (float*)(w + 10 * ACT + 884736);
    float* kinv     = (float*)(w + 10 * ACT + 890880);
    u16*   wqv_bf   = (u16*)(w + 10 * ACT + 897024);          // 384*192
    u16*   wk_bf    = (u16*)(w + 10 * ACT + 1044480);         // 192*192
    u16*   wproj_bf = (u16*)(w + 10 * ACT + 1118208);         // 192*192

    // 1) weight casts
    cvt_bf16<<<(73728 + 255) / 256, 256, 0, stream>>>(w_qv,   wqv_bf,   73728);
    cvt_bf16<<<(36864 + 255) / 256, 256, 0, stream>>>(w_k,    wk_bf,    36864);
    cvt_bf16<<<(36864 + 255) / 256, 256, 0, stream>>>(w_proj, wproj_bf, 36864);

    // 2) partial 3x3 conv + channel-last bf16 pack
    const int pcBlocks = (int)(((size_t)NTOT * CDIM) / 256);  // 98304
    pconv_pack<<<pcBlocks, 256, 0, stream>>>(x,   w_qv_pc, t_x);
    pconv_pack<<<pcBlocks, 256, 0, stream>>>(res, w_k_pc,  t_res);

    // 3) 1x1 conv GEMMs (M=192, K=192, N=131072), 64 rows of M per wave (MT=4)
    dim3 g1(NTOT / 16, CDIM / 64, 1);
    // Q (first 192 rows of w_qv) -> spatial-last [b*192][16384]
    gemm_wmma<0, 4><<<g1, 32, 0, stream>>>(wqv_bf, t_x, Qb,
        CDIM, CDIM, CDIM, CDIM, HW, HW, 0, 0, 0, 0, 0, 0);
    // V (rows 192..383) -> channel-last [b*n][192]
    gemm_wmma<1, 4><<<g1, 32, 0, stream>>>(wqv_bf + CDIM * CDIM, t_x, Vb,
        CDIM, CDIM, CDIM, CDIM, CDIM, 0, 0, 0, 0, 0, 0, 0);
    // K -> spatial-last
    gemm_wmma<0, 4><<<g1, 32, 0, stream>>>(wk_bf, t_res, Kb,
        CDIM, CDIM, CDIM, CDIM, HW, HW, 0, 0, 0, 0, 0, 0);

    // 4) row norms (1536 rows each)
    rownorm<<<BATCH * CDIM, 256, 0, stream>>>(Qb, qinv);
    rownorm<<<BATCH * CDIM, 256, 0, stream>>>(Kb, kinv);

    // 5) attention logits: per (b,h) 96x96, K=16384 (both operands K-major), MT=2
    dim3 g2(CHH / 16, CHH / 32, BATCH * HEADS);
    gemm_wmma<2, 2><<<g2, 32, 0, stream>>>(Qb, Kb, logits,
        CHH, HW, HW, HW, CHH, 0,
        (size_t)CDIM * HW, (size_t)CHH * HW,     // A strides (b, h)
        (size_t)CDIM * HW, (size_t)CHH * HW,     // B strides
        (size_t)2 * CHH * CHH, (size_t)CHH * CHH);

    // 6) softmax with fused L2-normalization scales + temperature
    softmax_rows<<<BATCH * HEADS * CHH, 128, 0, stream>>>(logits, qinv, kinv, temp, attnb);

    // 7) out = attn @ v : per (b,h) M=96, K=96, N=16384 -> channel-last (reuse t_x)
    dim3 g3(HW / 16, CHH / 32, BATCH * HEADS);
    gemm_wmma<1, 2><<<g3, 32, 0, stream>>>(attnb, Vb, t_x,
        CHH, CHH, CHH, CDIM, CDIM, 0,
        (size_t)2 * CHH * CHH, (size_t)CHH * CHH,
        (size_t)HW * CDIM,     (size_t)CHH,
        (size_t)HW * CDIM,     (size_t)CHH);

    // 8) final projection -> fp32 NCHW output (MT=4)
    gemm_wmma<3, 4><<<g1, 32, 0, stream>>>(wproj_bf, t_x, out,
        CDIM, CDIM, CDIM, CDIM, HW, HW, 0, 0, 0, 0, 0, 0);
}
